// KeyPointModelV2_55396488184311
// MI455X (gfx1250) — matrix operand order repaired and checked by reading, not compiled, not run on hardware
//
#include <hip/hip_runtime.h>
#include <math.h>

typedef __attribute__((ext_vector_type(16))) _Float16 v16h;
typedef __attribute__((ext_vector_type(8)))  float    v8f;
typedef __attribute__((ext_vector_type(4)))  unsigned int u32x4;
typedef __attribute__((ext_vector_type(8)))  int      i32x8;
typedef __attribute__((ext_vector_type(4)))  int      i32x4;

#define BATCH 8
#define HW    512
#define FHW   128
#define NPTS  11
#define COUT  256
#define KK    49      // 7x7 single-channel kernel
#define CASC  3

// ---- workspace layout (bytes) ----
#define WS_BSW   0          // 1024 _Float16 : pre-swizzled B fragments
#define WS_WE    2048       // 256*49 f32    : w_eff
#define WS_B2    52224      // 16 f32        : combined bias (padded)
#define WS_CRD   52288      // 176 f32       : coords scratch
#define WS_SCORE 53248      // 8*11*128*128 f32 : low-res score maps

// -----------------------------------------------------------------------------
// Kernel 1: fold weights.
//   w_eff[c][kk] = sum_i backbone_w[c][i][kk]
//   Wc[n][kk]    = sum_c fc_w[n][c] * w_eff[c][kk]   (11 x 49 combined conv)
//   b2[n]        = fc_b[n] + sum_c fc_w[n][c]*backbone_b[c]
//   Bsw          = Wc as WMMA f16 B-fragments (K x N = 64 x 16, 2 ksteps)
// -----------------------------------------------------------------------------
__global__ void __launch_bounds__(256) prep_kernel(
    const float* __restrict__ bw, const float* __restrict__ bb,
    const float* __restrict__ fcw, const float* __restrict__ fcb,
    _Float16* __restrict__ Bsw, float* __restrict__ wE, float* __restrict__ b2)
{
    __shared__ float Wc_sh[NPTS * KK];
    const int tid = threadIdx.x;

    {
        const int c = tid;
        for (int kk = 0; kk < KK; ++kk) {
            float s = bw[(c * 3 + 0) * KK + kk]
                    + bw[(c * 3 + 1) * KK + kk]
                    + bw[(c * 3 + 2) * KK + kk];
            wE[c * KK + kk] = s;
        }
    }
    __threadfence();
    __syncthreads();

    for (int idx = tid; idx < NPTS * KK; idx += 256) {
        int n = idx / KK, kk = idx % KK;
        float s = 0.f;
        for (int c = 0; c < COUT; ++c)
            s = fmaf(fcw[n * COUT + c], wE[c * KK + kk], s);
        Wc_sh[idx] = s;
    }

    if (tid < 16) {
        float s = 0.f;
        if (tid < NPTS) {
            s = fcb[tid];
            for (int c = 0; c < COUT; ++c)
                s = fmaf(fcw[tid * COUT + c], bb[c], s);
        }
        b2[tid] = s;
    }
    __syncthreads();

    // B-fragment swizzle: lane L -> N = L&15; kb = s*32 + (lanes 0-15: 0, lanes 16-31: 8);
    // element h<8 holds K=kb+h, element h>=8 holds K=kb+8+h.
    if (tid < 64) {
        int s = tid >> 5, lane = tid & 31;
        int n = lane & 15;
        int kb = s * 32 + ((lane < 16) ? 0 : 8);
        _Float16* dst = Bsw + (s * 32 + lane) * 16;
        for (int h = 0; h < 16; ++h) {
            int K = kb + ((h < 8) ? h : 8 + h);
            float v = (K < KK && n < NPTS) ? Wc_sh[n * KK + K] : 0.f;
            dst[h] = (_Float16)v;
        }
    }
}

// -----------------------------------------------------------------------------
// Kernel 2: fused conv(1ch,7x7,s4) -> 11 score channels via WMMA implicit GEMM.
// One block = one output row y of one image (128 px, 8 waves x 16-px M tiles).
// Input strip (7 rows x 512 f32) staged in LDS by the Tensor Data Mover.
// -----------------------------------------------------------------------------
__global__ void __launch_bounds__(256) conv_wmma_kernel(
    const float* __restrict__ images, const _Float16* __restrict__ Bsw,
    const float* __restrict__ b2, float* __restrict__ scores)
{
    __shared__ float smemf[7 * 512];     // raw f32 strip, row stride 512

    const int tid  = threadIdx.x;
    const int bi   = blockIdx.x >> 7;    // image
    const int y    = blockIdx.x & 127;   // output row
    const int wave = tid >> 5;
    const int lane = tid & 31;

    const float* img = images + (size_t)bi * HW * HW;
    const int iy0   = 4 * y - 3;
    const int ry0   = iy0 < 0 ? 0 : iy0;     // first in-image row
    const int roff  = ry0 - iy0;             // 0, or 3 when y==0
    const int nrows = 7 - roff;

    // sentinel-fill out-of-image top rows (only y==0 blocks): 0.5f -> 0 after norm
    if (roff > 0)
        for (int i = tid; i < roff * 512; i += 256) smemf[i] = 0.5f;

#if defined(__HIP_DEVICE_COMPILE__) && __has_builtin(__builtin_amdgcn_tensor_load_to_lds)
    if (wave == 0) {
        // Tensor DMA descriptor (D#), 2D tile: nrows x 512 f32, stride 512.
        unsigned long long ga = (unsigned long long)(const void*)(img + (size_t)ry0 * HW);
        unsigned int lds_base = (unsigned int)(unsigned long long)(void*)&smemf[roff * 512];
        u32x4 g0;
        g0[0] = 1u;                                             // count=1, user mode
        g0[1] = lds_base;                                       // lds_addr (bytes)
        g0[2] = (unsigned int)(ga & 0xffffffffull);             // global_addr lo
        g0[3] = (unsigned int)((ga >> 32) & 0x01ffffffull)
              | 0x80000000u;                                    // addr hi | type=2
        i32x8 g1;
        g1[0] = 0x00020000;                                     // data_size = 4B
        g1[1] = (int)(512u << 16);                              // tensor_dim0 = 512
        g1[2] = (int)(((unsigned)(HW - ry0) & 0xffffu) << 16);  // tensor_dim1
        g1[3] = (int)(512u << 16);                              // tile_dim0 = 512
        g1[4] = nrows;                                          // tile_dim1
        g1[5] = 512;                                            // dim0_stride lo
        g1[6] = 0;
        g1[7] = 0;
        i32x4 gz  = {0, 0, 0, 0};                               // <=2D: groups 2/3 unused
        i32x8 gz8 = {0, 0, 0, 0, 0, 0, 0, 0};
        __builtin_amdgcn_tensor_load_to_lds(g0, g1, gz, gz, gz8, 0);
        __builtin_amdgcn_s_wait_tensorcnt(0);
    }
#else
    for (int i = tid; i < nrows * 512; i += 256)
        smemf[roff * 512 + i] = img[(size_t)ry0 * HW + i];
#endif
    __syncthreads();

    const int x0 = wave * 16;
    const int m  = lane & 15;
    const int x  = x0 + m;

    v8f acc = {};
    #pragma unroll
    for (int s = 0; s < 2; ++s) {
        // A fragment (16x32 f16): lane<16: K={s*32+0..7, s*32+16..23};
        //                         lane>=16: K={s*32+8..15, s*32+24..31}
        const int k0 = s * 32 + ((lane < 16) ? 0 : 8);
        const int k1 = s * 32 + ((lane < 16) ? 16 : 24);
        v16h a;
        #pragma unroll
        for (int t = 0; t < 8; ++t) {
            int kk = k0 + t;
            if (kk < KK) {
                int cc = x * 4 - 3 + (kk % 7);
                float raw = (cc >= 0) ? smemf[(kk / 7) * 512 + cc] : 0.5f;
                a[t] = (_Float16)(raw - 0.5f);
            } else a[t] = (_Float16)0.0f;
            int kk2 = k1 + t;
            if (kk2 < KK) {
                int cc = x * 4 - 3 + (kk2 % 7);
                float raw = (cc >= 0) ? smemf[(kk2 / 7) * 512 + cc] : 0.5f;
                a[8 + t] = (_Float16)(raw - 0.5f);
            } else a[8 + t] = (_Float16)0.0f;
        }
        v16h bfrag = *(const v16h*)(Bsw + (s * 32 + lane) * 16);
        acc = __builtin_amdgcn_wmma_f32_16x16x32_f16(
            false, a, false, bfrag, (short)0, acc, false, false);
    }

    // D layout: VGPR j -> (lane<16: M=j, N=lane) / (lane>=16: M=8+j, N=lane-16)
    const int n = lane & 15;
    if (n < NPTS) {
        const float bias = b2[n];
        float* orow = scores + ((size_t)(bi * NPTS + n) * FHW + y) * FHW;
        const int mbase = (lane < 16) ? 0 : 8;
        #pragma unroll
        for (int j = 0; j < 8; ++j)
            orow[x0 + mbase + j] = acc[j] + bias;
    }
}

// -----------------------------------------------------------------------------
// Kernel 3: argmax over 512x512 align-corners bilinear upsample of scores.
// sigmoid is monotone -> argmax on interpolated scores directly.
// -----------------------------------------------------------------------------
__global__ void __launch_bounds__(256) argmax_kernel(
    const float* __restrict__ scores, float* __restrict__ crd)
{
    __shared__ float svals[256];
    __shared__ int   sidx[256];

    const int tid = threadIdx.x;
    const float* base = scores + (size_t)blockIdx.x * (FHW * FHW);
    const float scale = 127.0f / 511.0f;

    float best = -3.4e38f;
    int bidx = 0;
    for (int i = tid; i < HW * HW; i += 256) {
        int oy = i >> 9, ox = i & 511;
        float fy = oy * scale, fx = ox * scale;
        int y0 = (int)floorf(fy); int y1 = (y0 + 1 < FHW) ? y0 + 1 : FHW - 1;
        int xx0 = (int)floorf(fx); int xx1 = (xx0 + 1 < FHW) ? xx0 + 1 : FHW - 1;
        float wy = fy - (float)y0, wx = fx - (float)xx0;
        float v00 = base[y0 * FHW + xx0], v01 = base[y0 * FHW + xx1];
        float v10 = base[y1 * FHW + xx0], v11 = base[y1 * FHW + xx1];
        float v = (v00 * (1.f - wx) + v01 * wx) * (1.f - wy)
                + (v10 * (1.f - wx) + v11 * wx) * wy;
        if (v > best || (v == best && i < bidx)) { best = v; bidx = i; }
    }
    svals[tid] = best; sidx[tid] = bidx;
    __syncthreads();
    for (int s = 128; s > 0; s >>= 1) {
        if (tid < s) {
            float v2 = svals[tid + s]; int i2 = sidx[tid + s];
            if (v2 > svals[tid] || (v2 == svals[tid] && i2 < sidx[tid])) {
                svals[tid] = v2; sidx[tid] = i2;
            }
        }
        __syncthreads();
    }
    if (tid == 0) {
        int i = sidx[0];
        crd[blockIdx.x * 2 + 0] = (float)(i & 511);  // x (width)
        crd[blockIdx.x * 2 + 1] = (float)(i >> 9);   // y (height)
    }
}

// -----------------------------------------------------------------------------
// Kernel 4: 3 cascades; 256-dim point feature RECOMPUTED from w_eff + 7x7 patch.
// -----------------------------------------------------------------------------
__global__ void __launch_bounds__(256) cascade_kernel(
    const float* __restrict__ images, const float* __restrict__ wE,
    const float* __restrict__ bb, const float* __restrict__ hw,
    const float* __restrict__ hb, const float* __restrict__ crd,
    float* __restrict__ out)
{
    __shared__ float sc[2];
    __shared__ float patch[KK];
    __shared__ float r0[256], r1[256];

    const int tid = threadIdx.x;
    const int bp  = blockIdx.x;            // b*11 + p
    const int bi  = bp / NPTS;
    const float* img = images + (size_t)bi * HW * HW;

    if (tid == 0) { sc[0] = crd[bp * 2 + 0]; sc[1] = crd[bp * 2 + 1]; }
    __syncthreads();

    for (int it = 0; it < CASC; ++it) {
        float cy = sc[1], cx = sc[0];
        int hi = (int)rintf(cy * 0.25f); hi = hi < 0 ? 0 : (hi > FHW - 1 ? FHW - 1 : hi);
        int wi = (int)rintf(cx * 0.25f); wi = wi < 0 ? 0 : (wi > FHW - 1 ? FHW - 1 : wi);

        if (tid < KK) {
            int ky = tid / 7, kx = tid % 7;
            int iy = hi * 4 - 3 + ky, ix = wi * 4 - 3 + kx;
            float v = 0.f;
            if (iy >= 0 && iy < HW && ix >= 0 && ix < HW)
                v = img[iy * HW + ix] - 0.5f;
            patch[tid] = v;
        }
        __syncthreads();

        const int c = tid;
        float feat = bb[c];
        const float* wrow = wE + c * KK;
        #pragma unroll 7
        for (int kk = 0; kk < KK; ++kk)
            feat = fmaf(wrow[kk], patch[kk], feat);

        r0[tid] = feat * hw[(it * COUT + c) * 2 + 0];
        r1[tid] = feat * hw[(it * COUT + c) * 2 + 1];
        __syncthreads();
        for (int s = 128; s > 0; s >>= 1) {
            if (tid < s) { r0[tid] += r0[tid + s]; r1[tid] += r1[tid + s]; }
            __syncthreads();
        }
        if (tid == 0) {
            sc[0] += r0[0] + hb[it * 2 + 0];
            sc[1] += r1[0] + hb[it * 2 + 1];
        }
        __syncthreads();
    }
    if (tid == 0) {
        out[bp * 2 + 0] = sc[0];
        out[bp * 2 + 1] = sc[1];
    }
}

// -----------------------------------------------------------------------------
extern "C" void kernel_launch(void* const* d_in, const int* in_sizes, int n_in,
                              void* d_out, int out_size, void* d_ws, size_t ws_size,
                              hipStream_t stream) {
    (void)in_sizes; (void)n_in; (void)out_size; (void)ws_size;
    const float* images = (const float*)d_in[0];
    const float* bw     = (const float*)d_in[1];
    const float* bb     = (const float*)d_in[2];
    const float* fcw    = (const float*)d_in[3];
    const float* fcb    = (const float*)d_in[4];
    const float* hw     = (const float*)d_in[5];
    const float* hb     = (const float*)d_in[6];
    float* out = (float*)d_out;

    char* ws = (char*)d_ws;
    _Float16* Bsw  = (_Float16*)(ws + WS_BSW);
    float*    wE   = (float*)(ws + WS_WE);
    float*    b2   = (float*)(ws + WS_B2);
    float*    crd  = (float*)(ws + WS_CRD);
    float*    scr  = (float*)(ws + WS_SCORE);

    prep_kernel<<<1, 256, 0, stream>>>(bw, bb, fcw, fcb, Bsw, wE, b2);
    conv_wmma_kernel<<<BATCH * FHW, 256, 0, stream>>>(images, Bsw, b2, scr);
    argmax_kernel<<<BATCH * NPTS, 256, 0, stream>>>(scr, crd);
    cascade_kernel<<<BATCH * NPTS, 256, 0, stream>>>(images, wE, bb, hw, hb, crd, out);
}
